// FlashAttention_31628139167855
// MI455X (gfx1250) — matrix operand order repaired and checked by reading, compiled, not run on hardware
//
#include <hip/hip_runtime.h>
#include <hip/hip_bf16.h>

typedef __attribute__((ext_vector_type(16))) _Float16 v16h;
typedef __attribute__((ext_vector_type(8)))  float    v8f;
typedef __attribute__((ext_vector_type(4)))  _Float16 v4h;
typedef __attribute__((ext_vector_type(4)))  int      v4i;
typedef _Float16 h16;

#define GLOBAL_AS __attribute__((address_space(1)))
#define LDS_AS    __attribute__((address_space(3)))

// Pointer-to-int4 in the specific address spaces: the probe diagnostic showed
// __builtin_amdgcn_global_load_async_to_lds_b128 takes v4i* params.
typedef GLOBAL_AS v4i* gp128_t;
typedef LDS_AS    v4i* lp128_t;

#if defined(__has_builtin)
#  if __has_builtin(__builtin_amdgcn_global_load_async_to_lds_b128)
#    define HAVE_ASYNC_COPY 1
#  endif
#endif
#ifndef HAVE_ASYNC_COPY
#  define HAVE_ASYNC_COPY 0
#endif

namespace {

constexpr int kB = 2, kS = 2048, kD = 2048, kH = 16, kHD = 128;
constexpr float kScale = 0.08838834764831845f;  // 1/sqrt(128)

// 16-byte global -> LDS copy. On gfx1250 this is the ASYNCcnt-tracked
// GLOBAL_LOAD_ASYNC_TO_LDS_B128 data mover (no VGPR round-trip); guarded
// fallback is a synchronous b128 load + ds_store.
__device__ __forceinline__ void async_copy16(const h16* g, h16* l) {
#if HAVE_ASYNC_COPY
  __builtin_amdgcn_global_load_async_to_lds_b128((gp128_t)g, (lp128_t)l, 0, 0);
#else
  *(float4*)l = *(const float4*)g;
#endif
}

__device__ __forceinline__ void wait_async() {
#if HAVE_ASYNC_COPY
#  if __has_builtin(__builtin_amdgcn_s_wait_asynccnt)
  __builtin_amdgcn_s_wait_asynccnt(0);
#  else
  asm volatile("s_wait_asynccnt 0x0" ::: "memory");
#  endif
#endif
}

// ---- WMMA fragment helpers (layouts per CDNA5 ISA §7.12.2) ----
// A (16x32 f16, MxK): lane<16 -> row=lane,    K = {k0..k0+7, k0+16..k0+23}
//                     lane>=16 -> row=lane-16, K = {k0+8..k0+15, k0+24..k0+31}
__device__ __forceinline__ v16h frag_a(const h16* base, int row0, int k0, int ld) {
  const int lane = threadIdx.x & 31;
  const h16* p = base + (size_t)(row0 + (lane & 15)) * ld + k0 + ((lane >> 4) << 3);
  v16h f;
  ((float4*)&f)[0] = *(const float4*)p;
  ((float4*)&f)[1] = *(const float4*)(p + 16);
  return f;
}

// B (32x16 f16, KxN), source column-major: row `col` holds that output
// column's K values contiguously.
// lane<16 -> col=lane, k={k0..k0+7, k0+16..k0+23} ; lane>=16 -> col=lane-16, k={k0+8..k0+15, k0+24..k0+31}
__device__ __forceinline__ v16h frag_b(const h16* base, int col0, int k0, int ld) {
  const int lane = threadIdx.x & 31;
  const h16* p = base + (size_t)(col0 + (lane & 15)) * ld + k0 + ((lane >> 4) << 3);
  v16h f;
  ((float4*)&f)[0] = *(const float4*)p;
  ((float4*)&f)[1] = *(const float4*)(p + 16);
  return f;
}

__device__ __forceinline__ v8f wmma_f16(v16h a, v16h b, v8f c) {
  return __builtin_amdgcn_wmma_f32_16x16x32_f16(false, a, false, b, (short)0, c,
                                                false, false);
}

// ---------------- f32 -> f16 conversion (memory-trivial) ----------------
__global__ void cvt_f32_f16_kernel(const float* __restrict__ in,
                                   h16* __restrict__ out, int n4) {
  const int i = blockIdx.x * blockDim.x + threadIdx.x;
  if (i >= n4) return;
  const float4 f = ((const float4*)in)[i];
  v4h o;
  o[0] = (h16)f.x; o[1] = (h16)f.y; o[2] = (h16)f.z; o[3] = (h16)f.w;
  ((v4h*)out)[i] = o;
}

// ---------------- GEMM: C[M,N] = (X[M,K] @ W[N,K]^T + bias) * scale -------
// Block tile 256x128, 8 waves (4x2), wave tile 64x64 -> 16 accumulators.
// K staged 64-deep, double-buffered via async global->LDS copies:
// one s_wait_asynccnt + one barrier per stage. LDS pad 72 halves:
// 144B rows (16B aligned), 36*row mod 64 injective over 16 rows.
template <bool F16OUT>
__global__ __launch_bounds__(256) void gemm_xwt_kernel(
    const h16* __restrict__ X, const h16* __restrict__ W,
    const float* __restrict__ bias, void* __restrict__ out, float scale) {
  constexpr int LD = 72, KSTEP = 64, MT = 256, NT = 128;
  constexpr int STAGE = (MT + NT) * LD;  // halves per stage
  extern __shared__ h16 sm[];

  const int tid = threadIdx.x;
  const int lane = tid & 31;
  const int wid = tid >> 5;
  const int wm = (wid & 3) * 64;   // 4 wave-rows * 64
  const int wn = (wid >> 2) * 64;  // 2 wave-cols * 64
  const int m0 = blockIdx.y * MT;
  const int n0 = blockIdx.x * NT;

  // Per-thread copy assignment: A row = tid (64 halves), B row = tid>>1
  // (two threads per row, 32 halves each).
  const int brow = tid >> 1, bko = (tid & 1) << 5;

  auto issue_stage = [&](int stage, int k0) {
    h16* As = sm + stage * STAGE;
    h16* Bs = As + MT * LD;
    const h16* xs = X + (size_t)(m0 + tid) * kD + k0;
    h16* xd = As + tid * LD;
#pragma unroll
    for (int j = 0; j < 8; ++j) async_copy16(xs + j * 8, xd + j * 8);
    const h16* ws = W + (size_t)(n0 + brow) * kD + k0 + bko;
    h16* wd = Bs + brow * LD + bko;
#pragma unroll
    for (int j = 0; j < 4; ++j) async_copy16(ws + j * 8, wd + j * 8);
  };

  v8f acc[4][4] = {};

  issue_stage(0, 0);
  constexpr int NKT = kD / KSTEP;  // 32 stages
  for (int kt = 0; kt < NKT; ++kt) {
    wait_async();      // this wave's outstanding copies (tile kt) complete
    __syncthreads();   // -> tile kt visible block-wide; tile kt-1 fully read
    if (kt + 1 < NKT) issue_stage((kt + 1) & 1, (kt + 1) * KSTEP);

    const h16* As = sm + (kt & 1) * STAGE;
    const h16* Bs = As + MT * LD;
#pragma unroll
    for (int c = 0; c < 2; ++c) {
      v16h af[4], bf[4];
#pragma unroll
      for (int i = 0; i < 4; ++i) af[i] = frag_a(As, wm + i * 16, c * 32, LD);
#pragma unroll
      for (int j = 0; j < 4; ++j) bf[j] = frag_b(Bs, wn + j * 16, c * 32, LD);
#pragma unroll
      for (int i = 0; i < 4; ++i)
#pragma unroll
        for (int j = 0; j < 4; ++j) acc[i][j] = wmma_f16(af[i], bf[j], acc[i][j]);
    }
  }

  // Epilogue. C/D layout: row = r + 8*(lane>=16), col = lane&15.
  const int hi = lane >> 4, col = lane & 15;
#pragma unroll
  for (int i = 0; i < 4; ++i) {
#pragma unroll
    for (int j = 0; j < 4; ++j) {
      const int gn = n0 + wn + j * 16 + col;
      const float bv = bias[gn];
#pragma unroll
      for (int r = 0; r < 8; ++r) {
        const int gm = m0 + wm + i * 16 + hi * 8 + r;
        const float val = (acc[i][j][r] + bv) * scale;  // ref: (x@W^T+b)*scale
        if constexpr (F16OUT)
          ((h16*)out)[(size_t)gm * kD + gn] = (h16)val;
        else
          ((float*)out)[(size_t)gm * kD + gn] = val;
      }
    }
  }
}

// ---------------- Flash attention: softmax(Q K^T) V per (b, h) -----------
// Block: 128 threads = 4 waves; block owns 64 q-rows, wave owns 16.
// Q fragment resident in VGPRs across the whole key loop.
// LDS pad 136 halves: 272B rows (16B aligned), 4*row mod 64 injective
// over 16 rows -> conflict-free fragment reads.
__global__ __launch_bounds__(128) void attn_kernel(const h16* __restrict__ Q,
                                                   const h16* __restrict__ K,
                                                   const h16* __restrict__ V,
                                                   h16* __restrict__ O) {
  constexpr int LD = 136;
  extern __shared__ h16 smem[];
  h16* Ks = smem;            // [128 keys][LD]  row-major (d contiguous)
  h16* Vs = Ks + 128 * LD;   // [128 d][LD]     transposed (key contiguous)
  h16* Ps = Vs + 128 * LD;   // 4 waves x [16 q][LD] P scratch

  const int tid = threadIdx.x;
  const int lane = tid & 31;
  const int wid = tid >> 5;
  const int hi = lane >> 4, col = lane & 15;
  const int qblk = blockIdx.x;
  const int h = blockIdx.y;
  const int b = blockIdx.z;

  const size_t rowq = (size_t)(b * kS + qblk * 64 + wid * 16);

  // Resident, pre-scaled Q fragment: 16 rows x 128 d = 4 k-chunks.
  v16h qf[4];
  {
    const h16* qb = Q + (rowq + (lane & 15)) * kD + h * kHD;
#pragma unroll
    for (int c = 0; c < 4; ++c) {
      const h16* p = qb + c * 32 + (hi << 3);
      ((float4*)&qf[c])[0] = *(const float4*)p;
      ((float4*)&qf[c])[1] = *(const float4*)(p + 16);
    }
  }

  v8f oacc[8] = {};
  float mrow[8], lrow[8];
#pragma unroll
  for (int r = 0; r < 8; ++r) { mrow[r] = -__builtin_inff(); lrow[r] = 0.f; }

  h16* pw = Ps + wid * 16 * LD;

  for (int kb = 0; kb < kS / 128; ++kb) {
    // Stage K tile via async data mover; V tile transposed through VGPRs
    // (transpose can't be expressed as contiguous async copies).
    {
      const size_t krow = (size_t)(b * kS + kb * 128 + tid);
      const h16* ksrc = K + krow * kD + h * kHD;
      h16* kdst = Ks + tid * LD;
#pragma unroll
      for (int i = 0; i < 16; ++i) async_copy16(ksrc + i * 8, kdst + i * 8);
      const float4* vsrc = (const float4*)(V + krow * kD + h * kHD);
#pragma unroll
      for (int i = 0; i < 16; ++i) {
        const float4 f = vsrc[i];
        const h16* hp = (const h16*)&f;
#pragma unroll
        for (int j = 0; j < 8; ++j) Vs[(i * 8 + j) * LD + tid] = hp[j];
      }
    }
    wait_async();
    __syncthreads();

    // S = (Q*scale) K^T : 16 x 128, 8 key-tiles x 4 k-chunks = 32 WMMA.
    v8f sacc[8] = {};
#pragma unroll
    for (int t = 0; t < 8; ++t)
#pragma unroll
      for (int c = 0; c < 4; ++c)
        sacc[t] = wmma_f16(qf[c], frag_b(Ks, t * 16, c * 32, LD), sacc[t]);

    // Online softmax: each lane holds rows {r + 8*hi}; a row's 16 values of
    // each key-tile live across one 16-lane half, so xor 1..8 reduces a row.
    float rmax[8], rsum[8], alpha[8];
#pragma unroll
    for (int r = 0; r < 8; ++r) {
      float m = sacc[0][r];
#pragma unroll
      for (int t = 1; t < 8; ++t) m = fmaxf(m, sacc[t][r]);
#pragma unroll
      for (int off = 8; off >= 1; off >>= 1)
        m = fmaxf(m, __shfl_xor(m, off, 32));
      const float mnew = fmaxf(mrow[r], m);
      alpha[r] = __expf(mrow[r] - mnew);
      mrow[r] = mnew;
      rsum[r] = 0.f;
    }
#pragma unroll
    for (int t = 0; t < 8; ++t)
#pragma unroll
      for (int r = 0; r < 8; ++r) {
        const float p = __expf(sacc[t][r] - mrow[r]);
        sacc[t][r] = p;
        rsum[r] += p;
      }
#pragma unroll
    for (int r = 0; r < 8; ++r) {
#pragma unroll
      for (int off = 8; off >= 1; off >>= 1)
        rsum[r] += __shfl_xor(rsum[r], off, 32);
      lrow[r] = lrow[r] * alpha[r] + rsum[r];
    }
#pragma unroll
    for (int n = 0; n < 8; ++n)
#pragma unroll
      for (int r = 0; r < 8; ++r) oacc[n][r] *= alpha[r];

    // P (C layout) -> per-wave LDS scratch as f16 (A-layout source).
    // Same-wave LDS ops are in-order; no cross-wave hazard on pw.
#pragma unroll
    for (int t = 0; t < 8; ++t)
#pragma unroll
      for (int r = 0; r < 8; ++r)
        pw[(r + 8 * hi) * LD + t * 16 + col] = (h16)sacc[t][r];

    // O += P V : 8 d-tiles x 4 key-chunks = 32 WMMA.
#pragma unroll
    for (int n = 0; n < 8; ++n)
#pragma unroll
      for (int c = 0; c < 4; ++c)
        oacc[n] = wmma_f16(frag_a(pw, 0, c * 32, LD),
                           frag_b(Vs, n * 16, c * 32, LD), oacc[n]);
    __syncthreads();
  }

  // Normalize and store context (f16).
#pragma unroll
  for (int r = 0; r < 8; ++r) lrow[r] = 1.f / lrow[r];
#pragma unroll
  for (int n = 0; n < 8; ++n)
#pragma unroll
    for (int r = 0; r < 8; ++r)
      O[(rowq + 8 * hi + r) * kD + h * kHD + n * 16 + col] =
          (h16)(oacc[n][r] * lrow[r]);
}

}  // namespace

extern "C" void kernel_launch(void* const* d_in, const int* in_sizes, int n_in,
                              void* d_out, int out_size, void* d_ws,
                              size_t ws_size, hipStream_t stream) {
  const float* X = (const float*)d_in[0];
  const float* Wq = (const float*)d_in[1];
  const float* bq = (const float*)d_in[2];
  const float* Wk = (const float*)d_in[3];
  const float* bk = (const float*)d_in[4];
  const float* Wv = (const float*)d_in[5];
  const float* bv = (const float*)d_in[6];
  const float* Wo = (const float*)d_in[7];
  const float* bo = (const float*)d_in[8];

  // Workspace layout (f16 elements), total ~112 MB.
  const size_t nX = (size_t)kB * kS * kD;  // 8,388,608
  const size_t nW = (size_t)kD * kD;       // 4,194,304
  h16* Xh = (h16*)d_ws;
  h16* Wqh = Xh + nX;
  h16* Wkh = Wqh + nW;
  h16* Wvh = Wkh + nW;
  h16* Woh = Wvh + nW;
  h16* Qh = Woh + nW;
  h16* Kh = Qh + nX;
  h16* Vh = Kh + nX;
  h16* Ch = Vh + nX;

  // 1) f32 -> f16 (weights stay [n][k]: already the WMMA-B column-major form).
  cvt_f32_f16_kernel<<<dim3((unsigned)(nX / 4 / 256)), 256, 0, stream>>>(
      X, Xh, (int)(nX / 4));
  cvt_f32_f16_kernel<<<dim3((unsigned)(nW / 4 / 256)), 256, 0, stream>>>(
      Wq, Wqh, (int)(nW / 4));
  cvt_f32_f16_kernel<<<dim3((unsigned)(nW / 4 / 256)), 256, 0, stream>>>(
      Wk, Wkh, (int)(nW / 4));
  cvt_f32_f16_kernel<<<dim3((unsigned)(nW / 4 / 256)), 256, 0, stream>>>(
      Wv, Wvh, (int)(nW / 4));
  cvt_f32_f16_kernel<<<dim3((unsigned)(nW / 4 / 256)), 256, 0, stream>>>(
      Wo, Woh, (int)(nW / 4));

  // 2) QKV projections (SCALE folded into Q epilogue).
  const dim3 gg(kD / 128, (kB * kS) / 256);  // (16, 16)
  const size_t gsmem = (size_t)2 * (256 + 128) * 72 * sizeof(h16);  // ~108 KB
  gemm_xwt_kernel<true><<<gg, 256, gsmem, stream>>>(Xh, Wqh, bq, Qh, kScale);
  gemm_xwt_kernel<true><<<gg, 256, gsmem, stream>>>(Xh, Wkh, bk, Kh, 1.0f);
  gemm_xwt_kernel<true><<<gg, 256, gsmem, stream>>>(Xh, Wvh, bv, Vh, 1.0f);

  // 3) Attention.
  const dim3 ga(kS / 64, kH, kB);  // (32, 16, 2)
  const size_t asmem = (size_t)(128 * 136 + 128 * 136 + 64 * 136) * sizeof(h16);
  attn_kernel<<<ga, 128, asmem, stream>>>(Qh, Kh, Vh, Ch);

  // 4) Output projection, f32 result straight to d_out.
  gemm_xwt_kernel<false><<<gg, 256, gsmem, stream>>>(Ch, Woh, bo, d_out, 1.0f);
}
